// MambaBlock_3444563772118
// MI455X (gfx1250) — matrix operand order repaired and checked by reading, not compiled, not run on hardware
//
#include <hip/hip_runtime.h>
#include <hip/hip_bf16.h>
#include <math.h>

// ---------------------------------------------------------------------------
// Mamba block forward for MI455X (gfx1250, wave32, WMMA)
//   x:(2,2048,1024) -> out:(2,2048,1024), all fp32 in/out.
//   Big GEMMs run on v_wmma_f32_16x16x32_bf16 (f32 accumulate).
// ---------------------------------------------------------------------------

typedef __attribute__((ext_vector_type(16))) __bf16 v16bf;
typedef __attribute__((ext_vector_type(8)))  float  v8f;

#define DIMM    1024
#define DSTATE  16
#define DCONV   4
#define DINNER  2048
#define SEQ     2048
#define NBATCH  2
#define NROWS   (NBATCH * SEQ)       // 4096
#define NSSM    (2 * DSTATE + 1)     // 33

__device__ __forceinline__ float sigmoidf_(float x) { return 1.0f / (1.0f + __expf(-x)); }
__device__ __forceinline__ float siluf_(float x)    { return x * sigmoidf_(x); }
__device__ __forceinline__ float softplusf_(float x){ return (x > 20.0f) ? x : log1pf(__expf(x)); }

// ---------------------------------------------------------------------------
// WMMA GEMM:  C[M,N] = A[M,K] * W[N,K]^T   (both operands row-major fp32,
// converted on the fly to bf16; fp32 accumulate).
// Block: 256 threads = 8 waves; block tile 128x128, K-step 32.
// Wave grid 4(M) x 2(N); each wave computes 2x4 16x16 WMMA tiles (32x64).
// ---------------------------------------------------------------------------
#define BM 128
#define BN 128
#define BK 32
#define LDK (BK + 8)   // padded LDS stride (even -> 4B-aligned pair reads)

__global__ void __launch_bounds__(256)
wmma_gemm_bf16(const float* __restrict__ A, const float* __restrict__ W,
               float* __restrict__ C, int M, int N, int K)
{
    __shared__ __bf16 As[BM][LDK];
    __shared__ __bf16 Bs[BN][LDK];

    const int tid  = threadIdx.x;
    const int lane = tid & 31;
    const int l16  = lane & 15;
    const int half = lane >> 4;
    const int wid  = tid >> 5;     // 0..7
    const int wm   = wid & 3;      // 4 waves along M
    const int wn   = wid >> 2;     // 2 waves along N
    const int m0   = blockIdx.y * BM;
    const int n0   = blockIdx.x * BN;

    v8f acc[2][4] = {};

    for (int k0 = 0; k0 < K; k0 += BK) {
        // Stage A (BM x BK) and W (BN x BK) tiles into LDS as bf16.
        for (int i = tid; i < (BM * BK) / 2; i += 256) {
            int row = i >> 4;            // 16 float2-pairs per row
            int kp  = (i & 15) << 1;
            float2 fa = *(const float2*)(A + (size_t)(m0 + row) * K + k0 + kp);
            As[row][kp]     = (__bf16)fa.x;
            As[row][kp + 1] = (__bf16)fa.y;
            float2 fb = *(const float2*)(W + (size_t)(n0 + row) * K + k0 + kp);
            Bs[row][kp]     = (__bf16)fb.x;
            Bs[row][kp + 1] = (__bf16)fb.y;
        }
        __syncthreads();

        // Build fragments per ISA 16-bit A(16x32) / B(32x16) lane layouts.
        union { v16bf v; unsigned int u[8]; } af[2], bf[4];
        #pragma unroll
        for (int mi = 0; mi < 2; ++mi) {
            int row = wm * 32 + mi * 16 + l16;     // M = l16
            #pragma unroll
            for (int v = 0; v < 8; ++v) {
                int kb = ((v < 4) ? 0 : 16) + half * 8 + (v & 3) * 2;
                af[mi].u[v] = *(const unsigned int*)&As[row][kb];
            }
        }
        #pragma unroll
        for (int ni = 0; ni < 4; ++ni) {
            int row = wn * 64 + ni * 16 + l16;     // N = l16 (row of W = col of B)
            #pragma unroll
            for (int v = 0; v < 8; ++v) {
                int kb = ((v < 4) ? 0 : 16) + half * 8 + (v & 3) * 2;
                bf[ni].u[v] = *(const unsigned int*)&Bs[row][kb];
            }
        }
        #pragma unroll
        for (int mi = 0; mi < 2; ++mi)
            #pragma unroll
            for (int ni = 0; ni < 4; ++ni)
                acc[mi][ni] = __builtin_amdgcn_wmma_f32_16x16x32_bf16(
                    false, af[mi].v, false, bf[ni].v,
                    (short)0, acc[mi][ni], false, false);
        __syncthreads();
    }

    // C/D layout: VGPR v -> M = v + half*8, N = l16.
    #pragma unroll
    for (int mi = 0; mi < 2; ++mi)
        #pragma unroll
        for (int ni = 0; ni < 4; ++ni)
            #pragma unroll
            for (int v = 0; v < 8; ++v) {
                int row = m0 + wm * 32 + mi * 16 + half * 8 + v;
                int col = n0 + wn * 64 + ni * 16 + l16;
                C[(size_t)row * N + col] = acc[mi][ni][v];
            }
}

// ---------------------------------------------------------------------------
// Fused: depthwise causal conv(4) + SiLU + ssm = xc_silu @ W_x^T (33 outs).
// One block per (b,t) row; 256 threads x 8 channels each.
// ---------------------------------------------------------------------------
__global__ void __launch_bounds__(256)
conv_ssm_kernel(const float* __restrict__ xar, const float* __restrict__ conv_w,
                const float* __restrict__ conv_b, const float* __restrict__ Wx,
                float* __restrict__ xc, float* __restrict__ ssm)
{
    const int r    = blockIdx.x;
    const int b    = r >> 11;           // SEQ = 2048
    const int t    = r & (SEQ - 1);
    const int tid  = threadIdx.x;
    const int lane = tid & 31;
    const int wid  = tid >> 5;

    float v[8];
    #pragma unroll
    for (int j = 0; j < 8; ++j) {
        int c = tid + j * 256;
        float acc = conv_b[c];
        #pragma unroll
        for (int tap = 0; tap < DCONV; ++tap) {
            int tt = t - (DCONV - 1) + tap;
            if (tt >= 0)
                acc += conv_w[c * DCONV + tap] *
                       xar[(size_t)(b * SEQ + tt) * (2 * DINNER) + c];
        }
        float s = siluf_(acc);
        v[j] = s;
        xc[(size_t)r * DINNER + c] = s;
    }

    float part[NSSM];
    #pragma unroll
    for (int i = 0; i < NSSM; ++i) {
        float p = 0.f;
        #pragma unroll
        for (int j = 0; j < 8; ++j)
            p += Wx[i * DINNER + tid + j * 256] * v[j];
        part[i] = p;
    }

    __shared__ float red[NSSM * 8];
    #pragma unroll
    for (int i = 0; i < NSSM; ++i) {
        float p = part[i];
        #pragma unroll
        for (int off = 16; off > 0; off >>= 1)
            p += __shfl_down(p, off, 32);
        if (lane == 0) red[i * 8 + wid] = p;
    }
    __syncthreads();
    if (tid < NSSM) {
        float s = 0.f;
        #pragma unroll
        for (int w = 0; w < 8; ++w) s += red[tid * 8 + w];
        ssm[(size_t)r * NSSM + tid] = s;
    }
}

// ---------------------------------------------------------------------------
// Selective scan: one lane per (b, channel); h[16] lives in VGPRs.
// Fuses dt = softplus(dt_raw*W_dt+b_dt), state update, y, SiLU(y+res).
// Writes y2 in place over xc (same address read earlier this step).
// ---------------------------------------------------------------------------
__global__ void __launch_bounds__(128)
scan_kernel(const float* __restrict__ xar, const float* __restrict__ ssm,
            const float* __restrict__ Wdt, const float* __restrict__ bdt,
            const float* __restrict__ Alog, const float* __restrict__ Dv,
            float* __restrict__ xc /* in: xc_silu, out: y2 */)
{
    const int tid = threadIdx.x;
    const int b   = blockIdx.x >> 4;                 // 16 channel-blocks / batch
    const int c   = ((blockIdx.x & 15) << 7) + tid;  // 128 channels per block

    float Afac[DSTATE];
    #pragma unroll
    for (int s = 0; s < DSTATE; ++s) Afac[s] = -__expf(Alog[s]);  // A = -exp(A_log)
    float h[DSTATE];
    #pragma unroll
    for (int s = 0; s < DSTATE; ++s) h[s] = 0.f;

    const float wdt = Wdt[c];
    const float bd  = bdt[c];
    const float Dc  = Dv[c];

    __shared__ float sBC[NSSM];

    for (int t = 0; t < SEQ; ++t) {
        const size_t r = (size_t)b * SEQ + t;
        if (tid < NSSM) sBC[tid] = ssm[r * NSSM + tid];
        __syncthreads();

        float x   = xc[r * DINNER + c];
        float dtc = softplusf_(sBC[2 * DSTATE] * wdt + bd);
        float dtx = dtc * x;
        float y   = 0.f;
        #pragma unroll
        for (int s = 0; s < DSTATE; ++s) {
            float ad = __expf(Afac[s] * dtc);        // A_disc
            h[s] = ad * h[s] + dtx * sBC[s];         // B_t in sBC[0..15]
            y   += h[s] * sBC[DSTATE + s];           // C_t in sBC[16..31]
        }
        float u = y + Dc * x + xar[r * (2 * DINNER) + DINNER + c];  // + res
        xc[r * DINNER + c] = siluf_(u);
        __syncthreads();
    }
}

// ---------------------------------------------------------------------------
// Launcher. Workspace layout (bytes):
//   [0, 64MB)      x_and_res  (4096 x 4096 f32)
//   [64MB, 96MB)   xc_silu -> y2 in place (4096 x 2048 f32)
//   [96MB, +528KB) ssm (4096 x 33 f32)
// ---------------------------------------------------------------------------
extern "C" void kernel_launch(void* const* d_in, const int* in_sizes, int n_in,
                              void* d_out, int out_size, void* d_ws, size_t ws_size,
                              hipStream_t stream)
{
    (void)in_sizes; (void)n_in; (void)out_size; (void)ws_size;

    const float* x      = (const float*)d_in[0];
    const float* W_in   = (const float*)d_in[1];
    const float* conv_w = (const float*)d_in[2];
    const float* conv_b = (const float*)d_in[3];
    const float* W_x    = (const float*)d_in[4];
    const float* W_dt   = (const float*)d_in[5];
    const float* b_dt   = (const float*)d_in[6];
    const float* A_log  = (const float*)d_in[7];
    const float* Dvec   = (const float*)d_in[8];
    const float* W_out  = (const float*)d_in[9];
    float* out = (float*)d_out;

    char*  ws  = (char*)d_ws;
    float* xar = (float*)ws;
    float* xc  = (float*)(ws + (size_t)NROWS * 2 * DINNER * sizeof(float));
    float* ssm = (float*)(ws + (size_t)NROWS * 2 * DINNER * sizeof(float)
                             + (size_t)NROWS * DINNER * sizeof(float));

    // 1) x_and_res = x @ W_in^T     (M=4096, N=4096, K=1024)
    dim3 g1(2 * DINNER / BN, NROWS / BM);
    hipLaunchKernelGGL(wmma_gemm_bf16, g1, dim3(256), 0, stream,
                       x, W_in, xar, NROWS, 2 * DINNER, DIMM);

    // 2) conv + SiLU + W_x projection
    hipLaunchKernelGGL(conv_ssm_kernel, dim3(NROWS), dim3(256), 0, stream,
                       xar, conv_w, conv_b, W_x, xc, ssm);

    // 3) selective scan + SiLU(y + res), in place into xc
    hipLaunchKernelGGL(scan_kernel, dim3(32), dim3(128), 0, stream,
                       xar, ssm, W_dt, b_dt, A_log, Dvec, xc);

    // 4) out = y2 @ W_out^T         (M=4096, N=1024, K=2048)
    dim3 g2(DIMM / BN, NROWS / BM);
    hipLaunchKernelGGL(wmma_gemm_bf16, g2, dim3(256), 0, stream,
                       xc, W_out, out, NROWS, DIMM, DINNER);
}
